// DimeNet_19301583029079
// MI455X (gfx1250) — matrix operand order repaired and checked by reading, not compiled, not run on hardware
//
#include <hip/hip_runtime.h>
#include <hip/hip_bf16.h>
#include <math.h>

#define FDIM   128
#define NRBF   6
#define NSH    7
#define NBIL   8
#define NATOM  4096
#define NEDGE  65536
#define NTRI   262144
#define NBATCH 64
#define CUTF   5.0f
#define PI_F   3.14159265358979323846f
#define KEMB   288          /* 2*F + NRBF = 262 padded up to a multiple of 32 */
#define LDAEMB (KEMB + 8)

typedef __attribute__((ext_vector_type(16))) __bf16   v16bf;
typedef __attribute__((ext_vector_type(8)))  float    v8f;
typedef __attribute__((ext_vector_type(4)))  uint32_t u32x4;
typedef __attribute__((ext_vector_type(8)))  int      i32x8;
typedef __attribute__((ext_vector_type(4)))  int      i32x4;

__device__ __forceinline__ float swish_f(float x) { return x / (1.0f + expf(-x)); }

// ---------------------------------------------------------------------------
// WMMA fragment helpers (wave32, v_wmma_f32_16x16x32_bf16)
// A (16x32, MxK): lane L holds row M=L&15; elems e -> K=(L>>4)*16+e
// B (32x16, KxN): lane L holds col N=L&15; elems e -> K=(L>>4)*16+e
// C/D (16x16 f32): lane L col N=L&15, vgpr v row M=v+8*(L>>4)
// ---------------------------------------------------------------------------
__device__ __forceinline__ v16bf load_a_lds(const __bf16* lA, int stride, int k0, int lane) {
    int m    = lane & 15;
    int base = (lane >> 4) * 16;
    const __bf16* p = lA + m * stride + k0;
    v16bf a;
#pragma unroll
    for (int e = 0; e < 8; ++e) a[e] = p[base + e];
#pragma unroll
    for (int e = 0; e < 8; ++e) a[8 + e] = p[8 + base + e];
    return a;
}

__device__ __forceinline__ v16bf load_b_glb(const __bf16* B, int ldb, int k0, int n0, int lane) {
    int n  = lane & 15;
    int kb = k0 + (lane >> 4) * 16;
    const __bf16* p = B + (size_t)kb * ldb + n0 + n;
    v16bf b;
#pragma unroll
    for (int e = 0; e < 16; ++e) b[e] = p[(size_t)e * ldb];
    return b;
}

// B tile resident in LDS as [K rows][128 cols], row-major
__device__ __forceinline__ v16bf load_b_lds(const __bf16* lB, int k0, int n0loc, int lane) {
    int n  = lane & 15;
    int kb = k0 + (lane >> 4) * 16;
    const __bf16* p = lB + kb * 128 + n0loc + n;
    v16bf b;
#pragma unroll
    for (int e = 0; e < 16; ++e) b[e] = p[e * 128];
    return b;
}

// ---------------------------------------------------------------------------
// TDM: DMA a 2D bf16 tile (tile_dim0=128 cols x tile_dim1=K rows) from global
// (row stride = ldb elements) into LDS at lds_off.  ISA ch.8 D# layout:
//  g0: count=1 | lds_addr | global_addr(57b) | type=2
//  g1: data_size=1(2B) | tensor_dim0=ldb | tensor_dim1=K | tile_dim0=128 |
//      tile_dim1=K | tensor_dim0_stride=ldb
//  g2/g3/g4: zero (2D tensor, no iteration/gather)
// ---------------------------------------------------------------------------
__device__ __forceinline__ void tdm_load_tile_b128(const __bf16* gsrc, uint32_t lds_off,
                                                   int K, int ldb) {
    uint64_t ga = (uint64_t)(uintptr_t)gsrc;
    u32x4 g0;
    g0[0] = 1u;                                   // count=1, user mode
    g0[1] = lds_off;                              // LDS byte address
    g0[2] = (uint32_t)ga;                         // global_addr[31:0]
    g0[3] = (uint32_t)(ga >> 32) | (2u << 30);    // global_addr[56:32] | type=2
    i32x8 g1;
    g1[0] = 1 << 16;                              // data_size = 1 (2 bytes)
    g1[1] = (ldb & 0xffff) << 16;                 // tensor_dim0[15:0]
    g1[2] = ((ldb >> 16) & 0xffff) | ((K & 0xffff) << 16);  // dim0 hi | tensor_dim1
    g1[3] = 128 << 16;                            // tile_dim0 = 128
    g1[4] = K & 0xffff;                           // tile_dim1 = K
    g1[5] = ldb;                                  // tensor_dim0_stride
    g1[6] = 0;
    g1[7] = 0;
    i32x4 gz4 = {0, 0, 0, 0};
    i32x8 gz8 = {0, 0, 0, 0, 0, 0, 0, 0};
    __builtin_amdgcn_tensor_load_to_lds(g0, g1, gz4, gz4, gz8, 0);
}

// ---------------------------------------------------------------------------
// Weight conversion kernels
// ---------------------------------------------------------------------------
__global__ void cvt_f32_bf16(const float* __restrict__ s, __bf16* __restrict__ d, int n) {
    int i = blockIdx.x * blockDim.x + threadIdx.x;
    if (i < n) d[i] = (__bf16)s[i];
}

__global__ void build_wemb(const float* __restrict__ W, __bf16* __restrict__ d) {
    int i = blockIdx.x * blockDim.x + threadIdx.x;     // over KEMB*FDIM
    if (i >= KEMB * FDIM) return;
    int k = i / FDIM;
    d[i] = (k < 2 * FDIM + NRBF) ? (__bf16)W[i] : (__bf16)0.0f;
}

// ---------------------------------------------------------------------------
// Geometry: distances + radial Bessel basis
// ---------------------------------------------------------------------------
__global__ void edge_geom(const float* __restrict__ R, const int* __restrict__ ei,
                          const int* __restrict__ ej, float* __restrict__ Dij,
                          float* __restrict__ rbf) {
    int e = blockIdx.x * blockDim.x + threadIdx.x;
    if (e >= NEDGE) return;
    int i = ei[e], j = ej[e];
    float dx = R[i * 3 + 0] - R[j * 3 + 0];
    float dy = R[i * 3 + 1] - R[j * 3 + 1];
    float dz = R[i * 3 + 2] - R[j * 3 + 2];
    float d  = sqrtf(fmaxf(dx * dx + dy * dy + dz * dz, 0.0f));
    Dij[e] = d;
    float ds = fmaxf(d, 1e-6f);
    float c  = sqrtf(2.0f / CUTF);
#pragma unroll
    for (int n = 1; n <= NRBF; ++n)
        rbf[e * NRBF + n - 1] = c * sinf((float)n * PI_F * ds / CUTF) / ds;
}

// Triplet angles + spherical basis, pre-contracted with int_Wsbf for all 7 layers:
// sb_all[t, layer, b] = sum_s sbf[t,s] * Wsbf[layer, s, b]
__global__ void triplet_geom(const float* __restrict__ R, const float* __restrict__ Dij,
                             const int* __restrict__ ti, const int* __restrict__ tj,
                             const int* __restrict__ tk, const int* __restrict__ idexp,
                             const float* __restrict__ Wsbf, float* __restrict__ sb_all) {
    int t = blockIdx.x * blockDim.x + threadIdx.x;
    if (t >= NTRI) return;
    int ia = ti[t], ja = tj[t], ka = tk[t];
    float r1x = R[ja * 3 + 0] - R[ia * 3 + 0];
    float r1y = R[ja * 3 + 1] - R[ia * 3 + 1];
    float r1z = R[ja * 3 + 2] - R[ia * 3 + 2];
    float r2x = R[ka * 3 + 0] - R[ia * 3 + 0];
    float r2y = R[ka * 3 + 1] - R[ia * 3 + 1];
    float r2z = R[ka * 3 + 2] - R[ia * 3 + 2];
    float x  = r1x * r2x + r1y * r2y + r1z * r2z;
    float cx = r1y * r2z - r1z * r2y;
    float cy = r1z * r2x - r1x * r2z;
    float cz = r1x * r2y - r1y * r2x;
    float y  = sqrtf(cx * cx + cy * cy + cz * cz);
    float ang = atan2f(y, x);
    float dkj = fmaxf(Dij[idexp[t]], 1e-6f);
    float radial[NSH], angular[NSH];
#pragma unroll
    for (int n = 1; n <= NSH; ++n) radial[n - 1] = sinf((float)n * PI_F * dkj / CUTF) / dkj;
#pragma unroll
    for (int l = 0; l < NSH; ++l) angular[l] = cosf((float)l * ang);
    for (int L = 0; L < 7; ++L) {
#pragma unroll
        for (int b = 0; b < NBIL; ++b) {
            float s = 0.0f;
#pragma unroll
            for (int a = 0; a < NSH; ++a) {
                float acc = 0.0f;
#pragma unroll
                for (int r = 0; r < NSH; ++r)
                    acc += radial[r] * Wsbf[L * (NSH * NSH * NBIL) + (a * NSH + r) * NBIL + b];
                s += angular[a] * acc;
            }
            sb_all[(size_t)t * 56 + L * NBIL + b] = s;
        }
    }
}

// g[e,n] = sum_r rbf[e,r] * Wr[r,n]   (tiny K=6 projection)
__global__ void rbf_proj(const float* __restrict__ rbf, const float* __restrict__ Wr,
                         float* __restrict__ g) {
    int idx = blockIdx.x * blockDim.x + threadIdx.x;
    if (idx >= NEDGE * FDIM) return;
    int e = idx >> 7, n = idx & 127;
    float s = 0.0f;
#pragma unroll
    for (int r = 0; r < NRBF; ++r) s += rbf[e * NRBF + r] * Wr[r * FDIM + n];
    g[idx] = s;
}

// ---------------------------------------------------------------------------
// Embedding GEMM: m = swish([h_i | h_j | rbf] @ W_emb + b_emb)   K = 288 (padded)
// ---------------------------------------------------------------------------
__global__ void __launch_bounds__(256) emb_gemm(
    const int* __restrict__ Z, const int* __restrict__ ei, const int* __restrict__ ej,
    const float* __restrict__ emb, const float* __restrict__ rbf,
    const __bf16* __restrict__ B, const float* __restrict__ bias, float* __restrict__ m) {
    __shared__ __bf16 lA[16][LDAEMB];
    int tid = threadIdx.x, lane = tid & 31, w = tid >> 5;
    int row_base = blockIdx.x * 128;
    int n0 = w * 16;
    for (int mt = 0; mt < 8; ++mt) {
        int row0 = row_base + mt * 16;
        __syncthreads();
        for (int idx = tid; idx < 16 * KEMB; idx += 256) {
            int r = idx / KEMB, k = idx - r * KEMB;
            int e = row0 + r;
            float v;
            if (k < FDIM)                 v = emb[Z[ei[e]] * FDIM + k];
            else if (k < 2 * FDIM)        v = emb[Z[ej[e]] * FDIM + (k - FDIM)];
            else if (k < 2 * FDIM + NRBF) v = rbf[e * NRBF + (k - 2 * FDIM)];
            else                          v = 0.0f;
            lA[r][k] = (__bf16)v;
        }
        __syncthreads();
        v8f acc = {};
#pragma unroll
        for (int k0 = 0; k0 < KEMB; k0 += 32) {
            v16bf a = load_a_lds(&lA[0][0], LDAEMB, k0, lane);
            v16bf b = load_b_glb(B, FDIM, k0, n0, lane);
            acc = __builtin_amdgcn_wmma_f32_16x16x32_bf16(false, a, false, b,
                                                          (short)0, acc, false, false);
        }
        int colg  = n0 + (lane & 15);
        int rowg0 = row0 + ((lane >> 4) << 3);
#pragma unroll
        for (int v = 0; v < 8; ++v)
            m[(size_t)(rowg0 + v) * FDIM + colg] = swish_f(acc[v] + bias[colg]);
    }
}

// ---------------------------------------------------------------------------
// Generic K=128 WMMA GEMM: C = epilogue(A(+A2) @ B + bias)
// B tile (128 x 128-col chunk, bf16) is DMA'd to LDS by the Tensor Data Mover
// once per block; A row-tiles are staged via LDS with f32->bf16 conversion.
// mode bit0: swish | gmul!=null: multiply | resid!=null: residual add
// grid = (M/128, N/128), block = 256 (8 waves; wave w -> 16 columns)
// ---------------------------------------------------------------------------
__global__ void __launch_bounds__(256) wmma_gemm128(
    const float* __restrict__ A, const float* __restrict__ A2,
    const __bf16* __restrict__ B, const float* __restrict__ bias,
    const float* __restrict__ gmul, const float* __restrict__ resid,
    float* __restrict__ Cf, __bf16* __restrict__ Cbf,
    int M, int N, int mode) {
    const int K = 128;
    __shared__ __bf16 lB[K * 128];          // 32 KB weight tile (TDM destination)
    __shared__ __bf16 lA[16][K + 8];
    int tid = threadIdx.x, lane = tid & 31, w = tid >> 5;
    int row_base = blockIdx.x * 128;
    int ncol0 = blockIdx.y * 128;           // tile column origin in B
    int n0loc = w * 16;                     // this wave's columns within tile

    // --- async tensor DMA of the B tile into LDS (wave 0 issues once) ---
    if (w == 0) {
        tdm_load_tile_b128(B + ncol0, (uint32_t)(uintptr_t)(void*)&lB[0], K, N);
        __builtin_amdgcn_s_wait_tensorcnt(0);
    }
    __syncthreads();

    for (int mt = 0; mt < 8; ++mt) {
        int row0 = row_base + mt * 16;
        __syncthreads();
        for (int idx = tid; idx < 16 * K; idx += 256) {
            int r = idx >> 7, k = idx & 127;
            size_t gi = (size_t)(row0 + r) * K + k;
            float v = A[gi];
            if (A2) v += A2[gi];
            lA[r][k] = (__bf16)v;
        }
        __syncthreads();
        v8f acc = {};
#pragma unroll
        for (int k0 = 0; k0 < K; k0 += 32) {
            v16bf a = load_a_lds(&lA[0][0], K + 8, k0, lane);
            v16bf b = load_b_lds(&lB[0], k0, n0loc, lane);
            acc = __builtin_amdgcn_wmma_f32_16x16x32_bf16(false, a, false, b,
                                                          (short)0, acc, false, false);
        }
        int colg  = ncol0 + n0loc + (lane & 15);
        int rowg0 = row0 + ((lane >> 4) << 3);
#pragma unroll
        for (int v = 0; v < 8; ++v) {
            size_t o = (size_t)(rowg0 + v) * N + colg;
            float r = acc[v];
            if (bias) r += bias[colg];
            if (mode & 1) r = swish_f(r);
            if (gmul)  r *= gmul[o];
            if (resid) r += resid[o];
            if (Cf)  Cf[o]  = r;
            if (Cbf) Cbf[o] = (__bf16)r;
        }
    }
}

// ---------------------------------------------------------------------------
// Triplet combine: out[w,i] = sum_l sb[w,l] * z[kj(w), l*128+i], scatter-add
// into xkjsum[id_reduce[w], i].  z resides in L2 (134 MB bf16 < 192 MB L2).
// ---------------------------------------------------------------------------
__global__ void triplet_combine(const __bf16* __restrict__ z, const float* __restrict__ sb_all,
                                int layer, const int* __restrict__ idexp,
                                const int* __restrict__ idred, float* __restrict__ xkjsum) {
    int idx = blockIdx.x * blockDim.x + threadIdx.x;
    if (idx >= NTRI * FDIM) return;
    int t = idx >> 7, n = idx & 127;
    int esrc = idexp[t], edst = idred[t];
    const __bf16* zp = z + (size_t)esrc * (NBIL * FDIM) + n;
    const float*  sb = sb_all + (size_t)t * 56 + layer * NBIL;
    float s = 0.0f;
#pragma unroll
    for (int l = 0; l < NBIL; ++l) s += sb[l] * (float)zp[l * FDIM];
    atomicAdd(&xkjsum[(size_t)edst * FDIM + n], s);
}

// Output head: ta[idnb_i[e]] += m[e]*g[e]   (segment sum over atoms)
__global__ void out_edge_accum(const float* __restrict__ m, const float* __restrict__ g,
                               const int* __restrict__ ei, float* __restrict__ ta) {
    int idx = blockIdx.x * blockDim.x + threadIdx.x;
    if (idx >= NEDGE * FDIM) return;
    int e = idx >> 7, n = idx & 127;
    atomicAdd(&ta[(size_t)ei[e] * FDIM + n], m[idx] * g[idx]);
}

// P[a] += ta2[a,:] @ W2[:,0]
__global__ void out_final(const float* __restrict__ ta2, const float* __restrict__ W2,
                          float* __restrict__ P) {
    int a = blockIdx.x * blockDim.x + threadIdx.x;
    if (a >= NATOM) return;
    float s = 0.0f;
#pragma unroll 8
    for (int n = 0; n < FDIM; ++n) s += ta2[(size_t)a * FDIM + n] * W2[n];
    P[a] += s;
}

__global__ void batch_reduce(const float* __restrict__ P, const int* __restrict__ seg,
                             float* __restrict__ out) {
    int a = blockIdx.x * blockDim.x + threadIdx.x;
    if (a < NATOM) atomicAdd(&out[seg[a]], P[a]);
}

// ---------------------------------------------------------------------------
extern "C" void kernel_launch(void* const* d_in, const int* in_sizes, int n_in,
                              void* d_out, int out_size, void* d_ws, size_t ws_size,
                              hipStream_t stream) {
    (void)in_sizes; (void)n_in; (void)ws_size;
    const int*   Z        = (const int*)  d_in[0];
    const float* R        = (const float*)d_in[1];
    const int*   batchseg = (const int*)  d_in[2];
    const int*   ei       = (const int*)  d_in[3];
    const int*   ej       = (const int*)  d_in[4];
    const int*   idexp    = (const int*)  d_in[5];
    const int*   idred    = (const int*)  d_in[6];
    const int*   t_i      = (const int*)  d_in[7];
    const int*   t_j      = (const int*)  d_in[8];
    const int*   t_k      = (const int*)  d_in[9];
    const float* emb      = (const float*)d_in[10];
    const float* W_emb    = (const float*)d_in[11];
    const float* b_emb    = (const float*)d_in[12];
    const float* out_Wrbf = (const float*)d_in[13];
    const float* out_W1   = (const float*)d_in[14];
    const float* out_b1   = (const float*)d_in[15];
    const float* out_W2   = (const float*)d_in[16];
    const float* i_Wrbf   = (const float*)d_in[17];
    const float* i_Wsbf   = (const float*)d_in[18];
    const float* i_Wkj    = (const float*)d_in[19];
    const float* i_bkj    = (const float*)d_in[20];
    const float* i_Wji    = (const float*)d_in[21];
    const float* i_bji    = (const float*)d_in[22];
    const float* i_Wbil   = (const float*)d_in[23];
    const float* i_Wfin   = (const float*)d_in[24];
    const float* i_bfin   = (const float*)d_in[25];

    uint8_t* base = (uint8_t*)d_ws;
    size_t off = 0;
    auto carve = [&](size_t bytes) -> void* {
        void* p = base + off;
        off = (off + bytes + 255) & ~(size_t)255;
        return p;
    };
    float*  m      = (float*)carve((size_t)NEDGE * FDIM * 4);
    float*  xji    = (float*)carve((size_t)NEDGE * FDIM * 4);
    float*  xkj    = (float*)carve((size_t)NEDGE * FDIM * 4);
    float*  g      = (float*)carve((size_t)NEDGE * FDIM * 4);
    float*  xkjsum = (float*)carve((size_t)NEDGE * FDIM * 4);
    __bf16* zbuf   = (__bf16*)carve((size_t)NEDGE * NBIL * FDIM * 2);
    float*  dij    = (float*)carve((size_t)NEDGE * 4);
    float*  rbf    = (float*)carve((size_t)NEDGE * NRBF * 4);
    float*  sball  = (float*)carve((size_t)NTRI * 56 * 4);
    float*  ta     = (float*)carve((size_t)NATOM * FDIM * 4);
    float*  ta2    = (float*)carve((size_t)NATOM * FDIM * 4);
    float*  P      = (float*)carve((size_t)NATOM * 4);
    __bf16* bWemb  = (__bf16*)carve((size_t)KEMB * FDIM * 2);
    __bf16* bWji   = (__bf16*)carve((size_t)7 * FDIM * FDIM * 2);
    __bf16* bWkj   = (__bf16*)carve((size_t)7 * FDIM * FDIM * 2);
    __bf16* bWfin  = (__bf16*)carve((size_t)7 * FDIM * FDIM * 2);
    __bf16* bWbil  = (__bf16*)carve((size_t)7 * FDIM * NBIL * FDIM * 2);
    __bf16* bW1    = (__bf16*)carve((size_t)8 * FDIM * FDIM * 2);

    const int TB = 256;
    auto blks = [](long n) { return dim3((unsigned)((n + 255) / 256)); };

    // --- weight conversion (f32 -> bf16, L2-resident) ---
    cvt_f32_bf16<<<blks(7 * FDIM * FDIM), TB, 0, stream>>>(i_Wji, bWji, 7 * FDIM * FDIM);
    cvt_f32_bf16<<<blks(7 * FDIM * FDIM), TB, 0, stream>>>(i_Wkj, bWkj, 7 * FDIM * FDIM);
    cvt_f32_bf16<<<blks(7 * FDIM * FDIM), TB, 0, stream>>>(i_Wfin, bWfin, 7 * FDIM * FDIM);
    cvt_f32_bf16<<<blks(7L * FDIM * NBIL * FDIM), TB, 0, stream>>>(i_Wbil, bWbil, 7 * FDIM * NBIL * FDIM);
    cvt_f32_bf16<<<blks(8 * FDIM * FDIM), TB, 0, stream>>>(out_W1, bW1, 8 * FDIM * FDIM);
    build_wemb<<<blks(KEMB * FDIM), TB, 0, stream>>>(W_emb, bWemb);

    (void)hipMemsetAsync(P, 0, (size_t)NATOM * 4, stream);
    (void)hipMemsetAsync(d_out, 0, (size_t)out_size * sizeof(float), stream);

    // --- geometry ---
    edge_geom<<<blks(NEDGE), TB, 0, stream>>>(R, ei, ej, dij, rbf);
    triplet_geom<<<blks(NTRI), TB, 0, stream>>>(R, dij, t_i, t_j, t_k, idexp, i_Wsbf, sball);

    // --- embedding ---
    emb_gemm<<<dim3(NEDGE / 128), TB, 0, stream>>>(Z, ei, ej, emb, rbf, bWemb, b_emb, m);

    auto out_layer = [&](int k) {
        rbf_proj<<<blks((long)NEDGE * FDIM), TB, 0, stream>>>(rbf, out_Wrbf + (size_t)k * NRBF * FDIM, g);
        (void)hipMemsetAsync(ta, 0, (size_t)NATOM * FDIM * 4, stream);
        out_edge_accum<<<blks((long)NEDGE * FDIM), TB, 0, stream>>>(m, g, ei, ta);
        wmma_gemm128<<<dim3(NATOM / 128, 1), TB, 0, stream>>>(
            ta, nullptr, bW1 + (size_t)k * FDIM * FDIM, out_b1 + (size_t)k * FDIM,
            nullptr, nullptr, ta2, nullptr, NATOM, FDIM, 1);
        out_final<<<blks(NATOM), TB, 0, stream>>>(ta2, out_W2 + (size_t)k * FDIM, P);
    };

    out_layer(0);

    for (int i = 0; i < 7; ++i) {
        size_t wo = (size_t)i * FDIM * FDIM;
        // x_ji = swish(m @ Wji + bji)
        wmma_gemm128<<<dim3(NEDGE / 128, 1), TB, 0, stream>>>(
            m, nullptr, bWji + wo, i_bji + (size_t)i * FDIM,
            nullptr, nullptr, xji, nullptr, NEDGE, FDIM, 1);
        // g = rbf @ int_Wrbf[i]
        rbf_proj<<<blks((long)NEDGE * FDIM), TB, 0, stream>>>(rbf, i_Wrbf + (size_t)i * NRBF * FDIM, g);
        // x_kj = swish(m @ Wkj + bkj) * g
        wmma_gemm128<<<dim3(NEDGE / 128, 1), TB, 0, stream>>>(
            m, nullptr, bWkj + wo, i_bkj + (size_t)i * FDIM,
            g, nullptr, xkj, nullptr, NEDGE, FDIM, 1);
        // z = x_kj @ Wbil  (E x 1024, bf16 out, L2-resident)
        wmma_gemm128<<<dim3(NEDGE / 128, NBIL), TB, 0, stream>>>(
            xkj, nullptr, bWbil + (size_t)i * FDIM * NBIL * FDIM, nullptr,
            nullptr, nullptr, nullptr, zbuf, NEDGE, NBIL * FDIM, 0);
        // triplet bilinear combine + segment sum to edges
        (void)hipMemsetAsync(xkjsum, 0, (size_t)NEDGE * FDIM * 4, stream);
        triplet_combine<<<blks((long)NTRI * FDIM), TB, 0, stream>>>(zbuf, sball, i, idexp, idred, xkjsum);
        // m = m + swish((x_ji + xkjsum) @ Wfin + bfin)
        wmma_gemm128<<<dim3(NEDGE / 128, 1), TB, 0, stream>>>(
            xji, xkjsum, bWfin + wo, i_bfin + (size_t)i * FDIM,
            nullptr, m, m, nullptr, NEDGE, FDIM, 1);
        out_layer(i + 1);
    }

    batch_reduce<<<blks(NATOM), TB, 0, stream>>>(P, batchseg, (float*)d_out);
}
